// BernNet_16604343566804
// MI455X (gfx1250) — matrix operand order repaired and checked by reading, not compiled, not run on hardware
//
#include <hip/hip_runtime.h>
#include <hip/hip_bf16.h>
#include <stdint.h>

typedef __attribute__((ext_vector_type(16))) _Float16 v16h;
typedef __attribute__((ext_vector_type(8)))  float    v8f;
typedef __attribute__((ext_vector_type(2)))  _Float16 h2;

#define NNODES 100000
#define NEDGES 3200000
#define KORD   10
#define F_IN   512
#define F_HID  256
#define F_OUT  64

#define BM 32
#define BN 64
#define BK 32

// ---------------- gfx1250 async-to-LDS helpers ----------------

__device__ __forceinline__ unsigned lds_off(const void* p) {
  // generic LDS addresses carry the byte offset in the low 32 bits
  return (unsigned)(uintptr_t)p;
}

__device__ __forceinline__ void async_load_b128(unsigned lds_addr, const void* gptr) {
  asm volatile("global_load_async_to_lds_b128 %0, %1, off"
               :: "v"(lds_addr), "v"(gptr)
               : "memory");
}

__device__ __forceinline__ void wait_async0() {
#if __has_builtin(__builtin_amdgcn_s_wait_asynccnt)
  __builtin_amdgcn_s_wait_asynccnt(0);
#else
  asm volatile("s_wait_asynccnt 0x0" ::: "memory");
#endif
}

// ---------------- small utility kernels ----------------

__global__ void zero_kernel(float* __restrict__ p, int n) {
  int i = blockIdx.x * blockDim.x + threadIdx.x;
  if (i < n) p[i] = 0.0f;
}

__global__ void degree_kernel(const long long* __restrict__ ei, float* __restrict__ deg) {
  int e = blockIdx.x * blockDim.x + threadIdx.x;
  if (e < NEDGES) atomicAdd(&deg[(int)ei[e]], 1.0f);
}

__global__ void rsqrt_kernel(float* __restrict__ d) {
  int i = blockIdx.x * blockDim.x + threadIdx.x;
  if (i < NNODES) {
    float v = d[i];
    d[i] = (v > 0.0f) ? rsqrtf(v) : 0.0f;
  }
}

__global__ void edgew_kernel(const long long* __restrict__ ei, const float* __restrict__ dsq,
                             float* __restrict__ w) {
  int e = blockIdx.x * blockDim.x + threadIdx.x;
  if (e < NEDGES) w[e] = dsq[(int)ei[e]] * dsq[(int)ei[NEDGES + e]];
}

// beta[j] = scale * sum_i C(K,i)*relu(temp[i]) * [t^j] (1-t)^i (1+t)^(K-i)
__global__ void coeff_kernel(const float* __restrict__ temp, float* __restrict__ beta) {
  if (threadIdx.x != 0 || blockIdx.x != 0) return;
  const int K = KORD;
  float C[KORD + 1][KORD + 1];
  for (int i = 0; i <= K; ++i) {
    for (int j = 0; j <= K; ++j) C[i][j] = 0.0f;
    C[i][0] = 1.0f;
    for (int j = 1; j <= i; ++j)
      C[i][j] = (j == i) ? 1.0f : C[i - 1][j - 1] + C[i - 1][j];
  }
  const double scale = 1.0 / 1024.0;  // 1 / 2^K
  for (int j = 0; j <= K; ++j) {
    double s = 0.0;
    for (int i = 0; i <= K; ++i) {
      float T = temp[i] > 0.0f ? temp[i] : 0.0f;
      double g = 0.0;
      for (int p = 0; p <= i && p <= j; ++p) {
        int q = j - p;
        if (q > K - i) continue;
        double term = (double)C[i][p] * (double)C[K - i][q];
        g += (p & 1) ? -term : term;
      }
      s += (double)C[K][i] * (double)T * g;
    }
    beta[j] = (float)(scale * s);
  }
}

// Wt[n*Kc + k] = (f16) W[k*Nc + n]   (transpose + f32->f16)
__global__ void cvt_kernel(const float* __restrict__ W, _Float16* __restrict__ Wt,
                           int Kc, int Nc) {
  int idx = blockIdx.x * blockDim.x + threadIdx.x;
  if (idx < Kc * Nc) {
    int k = idx / Nc, n = idx % Nc;
    Wt[(size_t)n * Kc + k] = (_Float16)W[idx];
  }
}

// ---------------- WMMA GEMM kernels ----------------
// Block tile BM x BN (32x64), 8 waves each owning a 16x16 C-tile.
// A/B tiles staged in LDS via global_load_async_to_lds_b128 (ASYNCcnt path),
// fragments built from LDS (ds loads), MAC via v_wmma_f32_16x16x32_f16.

__global__ __launch_bounds__(256)
void gemm1_wmma(const float* __restrict__ A,      // [M, 512] f32
                const _Float16* __restrict__ Bt,  // [256, 512] f16 (W1^T)
                const float* __restrict__ bias,   // [256]
                _Float16* __restrict__ H) {       // [M, 256] f16, relu
  const int Kc = F_IN;
  __shared__ float    As[BM * BK];      // 4 KB, [row][k]
  __shared__ _Float16 Bs[BN * BK];      // 4 KB, [n][k]

  int tid = threadIdx.x;
  int wid = tid >> 5, lane = tid & 31;
  int l15 = lane & 15, hi = lane >> 4;
  int wm = wid >> 2, wn = wid & 3;       // 2 x 4 wave grid
  int bm = blockIdx.x;                   // 3125 row tiles (exact)
  int bn = blockIdx.y;                   // 4 col tiles of 64

  unsigned asBase = lds_off(As);
  unsigned bsBase = lds_off(Bs);

  // per-thread staging chunk assignments (one b128 each per tile)
  int ar = tid >> 3, ac = tid & 7;       // A: 32 rows x 8 chunks of 4 floats
  int brn = tid >> 2, bc = tid & 3;      // B: 64 n-rows x 4 chunks of 8 halves
  const float*    Ag = A  + (size_t)(bm * BM + ar) * Kc + ac * 4;
  const _Float16* Bg = Bt + (size_t)(bn * BN + brn) * Kc + bc * 8;
  unsigned asDst = asBase + (unsigned)(ar * BK + ac * 4) * 4u;
  unsigned bsDst = bsBase + (unsigned)(brn * BK + bc * 8) * 2u;

  v8f acc = {};
  for (int kk = 0; kk < Kc; kk += BK) {
    async_load_b128(asDst, Ag + kk);
    async_load_b128(bsDst, Bg + kk);
    wait_async0();
    __syncthreads();

    v16h a, b;
    const float* Ar = As + (wm * 16 + l15) * BK;
#pragma unroll
    for (int j = 0; j < 8; ++j) {  // A 16x32 f16 fragment, contiguous k = 16*hi + s
      int kb = hi * 16 + 2 * j;
      float2 f = *(const float2*)(Ar + kb);
      a[2 * j]     = (_Float16)f.x;
      a[2 * j + 1] = (_Float16)f.y;
    }
    const _Float16* Br = Bs + (wn * 16 + l15) * BK + hi * 16;
#pragma unroll
    for (int j = 0; j < 8; ++j) {
      h2 bb = *(const h2*)(Br + 2 * j);
      b[2 * j]     = bb.x;
      b[2 * j + 1] = bb.y;
    }
    acc = __builtin_amdgcn_wmma_f32_16x16x32_f16(false, a, false, b,
                                                 (short)0, acc, false, false);
    __syncthreads();
  }

  int n = bn * BN + wn * 16 + l15;
  float bv = bias[n];
#pragma unroll
  for (int j = 0; j < 8; ++j) {
    int m = bm * BM + wm * 16 + hi * 8 + j;
    float v = acc[j] + bv;
    H[(size_t)m * F_HID + n] = (_Float16)(v > 0.0f ? v : 0.0f);
  }
}

__global__ __launch_bounds__(256)
void gemm2_wmma(const _Float16* __restrict__ A,   // [M, 256] f16
                const _Float16* __restrict__ Bt,  // [64, 256] f16 (W2^T)
                const float* __restrict__ bias,   // [64]
                float* __restrict__ X) {          // [M, 64] f32
  const int Kc = F_HID;
  __shared__ _Float16 As[BM * BK];      // 2 KB, [row][k]
  __shared__ _Float16 Bs[BN * BK];      // 4 KB, [n][k]

  int tid = threadIdx.x;
  int wid = tid >> 5, lane = tid & 31;
  int l15 = lane & 15, hi = lane >> 4;
  int wm = wid >> 2, wn = wid & 3;
  int bm = blockIdx.x;                   // 3125 row tiles (exact)

  unsigned asBase = lds_off(As);
  unsigned bsBase = lds_off(Bs);

  // A: 32 rows x 4 chunks of 8 halves = 128 chunks (threads 0..127 stage)
  int ar = tid >> 2, ac = tid & 3;
  // B: 64 n-rows x 4 chunks of 8 halves = 256 chunks
  int brn = tid >> 2, bc = tid & 3;
  const _Float16* Ag = A  + (size_t)(bm * BM + (ar & 31)) * Kc + ac * 8;
  const _Float16* Bg = Bt + (size_t)brn * Kc + bc * 8;
  unsigned asDst = bsBase;  // placeholder, set below
  asDst = asBase + (unsigned)((ar & 31) * BK + ac * 8) * 2u;
  unsigned bsDst = bsBase + (unsigned)(brn * BK + bc * 8) * 2u;

  v8f acc = {};
  for (int kk = 0; kk < Kc; kk += BK) {
    if (tid < 128) async_load_b128(asDst, Ag + kk);
    async_load_b128(bsDst, Bg + kk);
    wait_async0();
    __syncthreads();

    v16h a, b;
    const _Float16* Ar = As + (wm * 16 + l15) * BK;
#pragma unroll
    for (int j = 0; j < 8; ++j) {
      int kb = hi * 16 + 2 * j;
      h2 aa = *(const h2*)(Ar + kb);
      a[2 * j]     = aa.x;
      a[2 * j + 1] = aa.y;
    }
    const _Float16* Br = Bs + (wn * 16 + l15) * BK + hi * 16;
#pragma unroll
    for (int j = 0; j < 8; ++j) {
      h2 bb = *(const h2*)(Br + 2 * j);
      b[2 * j]     = bb.x;
      b[2 * j + 1] = bb.y;
    }
    acc = __builtin_amdgcn_wmma_f32_16x16x32_f16(false, a, false, b,
                                                 (short)0, acc, false, false);
    __syncthreads();
  }

  int n = wn * 16 + l15;                 // Nc == BN == 64, single n-block
  float bv = bias[n];
#pragma unroll
  for (int j = 0; j < 8; ++j) {
    int m = bm * BM + wm * 16 + hi * 8 + j;
    X[(size_t)m * F_OUT + n] = acc[j] + bv;
  }
}

// ---------------- SpMM: vn += w_e * v[col]  (one wave per edge) ----------------

__global__ void spmm_kernel(const long long* __restrict__ ei, const float* __restrict__ w,
                            const float* __restrict__ v, float* __restrict__ vn) {
  int wave = (blockIdx.x * blockDim.x + threadIdx.x) >> 5;
  if (wave >= NEDGES) return;
  int lane = threadIdx.x & 31;
  int r = (int)ei[wave];
  int c = (int)ei[NEDGES + wave];
  float we = w[wave];
  float2 vv = *(const float2*)(v + (size_t)c * F_OUT + lane * 2);
  float* dst = vn + (size_t)r * F_OUT + lane * 2;
  atomicAdd(dst,     we * vv.x);
  atomicAdd(dst + 1, we * vv.y);
}

// out = (accum ? out : 0) + beta[j] * v
__global__ void axpy_kernel(float* __restrict__ out, const float* __restrict__ v,
                            const float* __restrict__ beta, int j, int accum) {
  int i = blockIdx.x * blockDim.x + threadIdx.x;
  if (i < NNODES * F_OUT) {
    float b = beta[j];
    out[i] = (accum ? out[i] : 0.0f) + b * v[i];
  }
}

// ---------------- host-side launch ----------------

extern "C" void kernel_launch(void* const* d_in, const int* in_sizes, int n_in,
                              void* d_out, int out_size, void* d_ws, size_t ws_size,
                              hipStream_t stream) {
  (void)in_sizes; (void)n_in; (void)out_size; (void)ws_size;

  const float*     node_feat = (const float*)d_in[0];
  const long long* ei        = (const long long*)d_in[1];   // int64 edge_index [2, E]
  const float*     W1        = (const float*)d_in[2];
  const float*     b1        = (const float*)d_in[3];
  const float*     W2        = (const float*)d_in[4];
  const float*     b2        = (const float*)d_in[5];
  const float*     temp      = (const float*)d_in[6];
  float* out = (float*)d_out;

  char* ws = (char*)d_ws;
  size_t off = 0;
  auto alloc = [&](size_t bytes) -> char* {
    char* p = ws + off;
    off = (off + bytes + 255) & ~(size_t)255;
    return p;
  };

  float*     deg  = (float*)alloc((size_t)NNODES * 4);              // reused as dsq
  float*     wbuf = (float*)alloc((size_t)NEDGES * 4);
  float*     beta = (float*)alloc((KORD + 1) * 4);
  _Float16*  W1t  = (_Float16*)alloc((size_t)F_IN * F_HID * 2);
  _Float16*  W2t  = (_Float16*)alloc((size_t)F_HID * F_OUT * 2);
  _Float16*  H    = (_Float16*)alloc((size_t)NNODES * F_HID * 2);
  float*     vA   = (float*)alloc((size_t)NNODES * F_OUT * 4);
  float*     vB   = (float*)alloc((size_t)NNODES * F_OUT * 4);

  const int T = 256;
  const int NT = NNODES * F_OUT;

  // normalization weights
  zero_kernel  <<<(NNODES + T - 1) / T, T, 0, stream>>>(deg, NNODES);
  degree_kernel<<<(NEDGES + T - 1) / T, T, 0, stream>>>(ei, deg);
  rsqrt_kernel <<<(NNODES + T - 1) / T, T, 0, stream>>>(deg);
  edgew_kernel <<<(NEDGES + T - 1) / T, T, 0, stream>>>(ei, deg, wbuf);

  // polynomial coefficients beta[0..K] from temp
  coeff_kernel<<<1, 32, 0, stream>>>(temp, beta);

  // weights -> f16 transposed
  cvt_kernel<<<(F_IN * F_HID + T - 1) / T, T, 0, stream>>>(W1, W1t, F_IN, F_HID);
  cvt_kernel<<<(F_HID * F_OUT + T - 1) / T, T, 0, stream>>>(W2, W2t, F_HID, F_OUT);

  // MLP via WMMA (LDS-staged, async loads)
  gemm1_wmma<<<dim3(NNODES / BM, F_HID / BN), T, 0, stream>>>(node_feat, W1t, b1, H);
  gemm2_wmma<<<dim3(NNODES / BM, 1), T, 0, stream>>>(H, W2t, b2, vA);

  // out = sum_j beta[j] * A_hat^j x  (10 SpMM sweeps instead of 65)
  axpy_kernel<<<(NT + T - 1) / T, T, 0, stream>>>(out, vA, beta, 0, 0);
  float* src = vA;
  float* dst = vB;
  for (int j = 1; j <= KORD; ++j) {
    zero_kernel<<<(NT + T - 1) / T, T, 0, stream>>>(dst, NT);
    spmm_kernel<<<NEDGES / 8, T, 0, stream>>>(ei, wbuf, src, dst);
    axpy_kernel<<<(NT + T - 1) / T, T, 0, stream>>>(out, dst, beta, j, 1);
    float* t = src; src = dst; dst = t;
  }
}
